// TRAlign_68401649156513
// MI455X (gfx1250) — matrix operand order repaired and checked by reading, not compiled, not run on hardware
//
#include <hip/hip_runtime.h>
#include <hip/hip_bf16.h>
#include <cstdint>

typedef __bf16 bf16_t;
typedef __attribute__((ext_vector_type(16))) __bf16 v16bf;
typedef __attribute__((ext_vector_type(8)))  float  v8f;

namespace {

constexpr int NTOK  = 4096;            // target rows (N)
constexpr int MTOK  = 4096;            // source rows (M)
constexpr int RDIM  = 144;             // region dim
constexpr int LDIM  = 3584;            // llm dim
constexpr int NHEAD = 4;
constexpr int HDIM  = 36;              // head dim
constexpr int VHDIM = 896;             // llm dim per head
constexpr int HPAD  = 64;              // head dim padded for WMMA K
constexpr int QKLD  = NHEAD * HPAD;    // 256: padded Q/K row length
constexpr int RPAD  = 160;             // region dim padded to mult of 32

// ---------------- workspace layout (all offsets 256B aligned) ----------------
constexpr size_t OFF_TA   = 0;                                      // target bf16 [4096 x 160]
constexpr size_t OFF_SA   = OFF_TA   + (size_t)NTOK * RPAD * 2;     // source bf16 [4096 x 3584]
constexpr size_t OFF_VA   = OFF_SA   + (size_t)MTOK * LDIM * 2;     // value  bf16 [4096 x 3584]
constexpr size_t OFF_WQT  = OFF_VA   + (size_t)MTOK * LDIM * 2;     // wq^T   [144 x 160]
constexpr size_t OFF_WKT  = OFF_WQT  + (size_t)RDIM * RPAD * 2;     // wk^T   [144 x 3584]
constexpr size_t OFF_WVT  = OFF_WKT  + (size_t)RDIM * LDIM * 2;     // wv^T   [4][896 x 3584]
constexpr size_t OFF_RST  = OFF_WVT  + (size_t)NHEAD * VHDIM * LDIM * 2; // resid^T [3584 x 160]
constexpr size_t OFF_OUTT = OFF_RST  + (size_t)LDIM * RPAD * 2;     // out^T  [3584 x 3584]
constexpr size_t OFF_QP   = OFF_OUTT + (size_t)LDIM * LDIM * 2;     // Q padded [4096 x 256]
constexpr size_t OFF_KP   = OFF_QP   + (size_t)NTOK * QKLD * 2;     // K padded [4096 x 256]
constexpr size_t OFF_ATTN = OFF_KP   + (size_t)MTOK * QKLD * 2;     // per-head scores/attn [4096x4096] bf16
constexpr size_t OFF_VT   = OFF_ATTN + (size_t)NTOK * MTOK * 2;     // V^T [4][896 x 4096] bf16
constexpr size_t OFF_X    = OFF_VT   + (size_t)NHEAD * VHDIM * MTOK * 2; // x f32 [4096 x 3584]
constexpr size_t OFF_Y    = OFF_X    + (size_t)NTOK * LDIM * 4;     // y bf16 [4096 x 3584]

union Frag16 { v16bf v; uint4 q[2]; };

__device__ inline v8f zero_v8f() {
  v8f z;
  #pragma unroll
  for (int i = 0; i < 8; ++i) z[i] = 0.0f;
  return z;
}

// A fragment, 16x32 bf16, row-major A. Lane l holds row m=l&15.
// elems 0..7 -> k = k0 + half*8 + j ; elems 8..15 -> k = k0 + 16 + half*8 + (j-8)
__device__ inline v16bf load_a_frag(const bf16_t* __restrict__ A, int lda,
                                    int m, int k0, int lane) {
  int half8 = (lane >> 4) << 3;
  const bf16_t* p = A + (size_t)m * lda + k0 + half8;
  Frag16 f;
  f.q[0] = *reinterpret_cast<const uint4*>(p);
  f.q[1] = *reinterpret_cast<const uint4*>(p + 16);
  return f.v;
}

// B fragment, 32x16 bf16 from B^T (row n holds K contiguous).
// Lane l holds col n=l&15; elems 0..7 -> k = k0 + (l>>4)*8 + j ; elems 8..15 -> k = k0 + 16 + (l>>4)*8 + (j-8).
__device__ inline v16bf load_b_frag(const bf16_t* __restrict__ Bt, int ldb,
                                    int n, int k0, int lane) {
  int half16 = (lane >> 4) << 3;
  const bf16_t* p = Bt + (size_t)n * ldb + k0 + half16;
  Frag16 f;
  f.q[0] = *reinterpret_cast<const uint4*>(p);
  f.q[1] = *reinterpret_cast<const uint4*>(p + 16);
  return f.v;
}

// Wave tile: (MT*16) x (NT*16). Block = 8 waves stacked along M.
// MODE 0: f32 row-major store (optional accumulate)
// MODE 1: bf16 row-major store
// MODE 2: bf16 transposed store (C[n*ldc + m])
// MODE 3: bf16 Q/K store, col c -> (c/36)*64 + c%36 into row stride 256
template <int MT, int NT, int MODE>
__global__ __launch_bounds__(256) void gemm_wmma(
    const bf16_t* __restrict__ A, const bf16_t* __restrict__ Bt,
    const float* __restrict__ bias, void* __restrict__ Cout,
    int K, int lda, int ldb, int ldc, float alpha, int accum) {
  const int wave = threadIdx.x >> 5;
  const int lane = threadIdx.x & 31;
  const int m_base = (blockIdx.x * 8 + wave) * (MT * 16);
  const int n_base = blockIdx.y * (NT * 16);
  const int mrow = m_base + (lane & 15);
  const int ncol = lane & 15;

  v8f acc[MT][NT];
  #pragma unroll
  for (int im = 0; im < MT; ++im)
    #pragma unroll
    for (int t = 0; t < NT; ++t) acc[im][t] = zero_v8f();

  for (int k0 = 0; k0 < K; k0 += 32) {
    v16bf a[MT];
    #pragma unroll
    for (int im = 0; im < MT; ++im)
      a[im] = load_a_frag(A, lda, mrow + im * 16, k0, lane);
    #pragma unroll
    for (int t = 0; t < NT; ++t) {
      v16bf b = load_b_frag(Bt, ldb, n_base + t * 16 + ncol, k0, lane);
      #pragma unroll
      for (int im = 0; im < MT; ++im)
        acc[im][t] = __builtin_amdgcn_wmma_f32_16x16x32_bf16(
            false, a[im], false, b, (short)0, acc[im][t], false, false);
    }
  }

  const int lrow = (lane >> 4) << 3;  // 0 or 8
  #pragma unroll
  for (int im = 0; im < MT; ++im) {
    #pragma unroll
    for (int t = 0; t < NT; ++t) {
      const int n = n_base + t * 16 + ncol;
      const float bv = bias ? bias[n] : 0.0f;
      #pragma unroll
      for (int r = 0; r < 8; ++r) {
        const int mm = m_base + im * 16 + lrow + r;
        float val = alpha * (acc[im][t][r] + bv);
        if (MODE == 0) {
          float* C = reinterpret_cast<float*>(Cout);
          size_t idx = (size_t)mm * ldc + n;
          if (accum) val += C[idx];
          C[idx] = val;
        } else if (MODE == 1) {
          bf16_t* C = reinterpret_cast<bf16_t*>(Cout);
          C[(size_t)mm * ldc + n] = (bf16_t)val;
        } else if (MODE == 2) {
          bf16_t* C = reinterpret_cast<bf16_t*>(Cout);
          C[(size_t)n * ldc + mm] = (bf16_t)val;
        } else {
          bf16_t* C = reinterpret_cast<bf16_t*>(Cout);
          int head = n / HDIM, d = n % HDIM;
          C[(size_t)mm * QKLD + head * HPAD + d] = (bf16_t)val;
        }
      }
    }
  }
}

// f32 -> bf16, pad cols to ldd with zeros
__global__ __launch_bounds__(256) void cvt_pad(const float* __restrict__ src,
                                               bf16_t* __restrict__ dst,
                                               int cols, int ldd, size_t total) {
  for (size_t i = (size_t)blockIdx.x * 256 + threadIdx.x; i < total;
       i += (size_t)gridDim.x * 256) {
    int r = (int)(i / ldd), c = (int)(i % ldd);
    dst[i] = (c < cols) ? (bf16_t)src[(size_t)r * cols + c] : (bf16_t)0.0f;
  }
}

// f32 [K x N] -> bf16 B^T [N x ldd], pad K to ldd with zeros
__global__ __launch_bounds__(256) void cvt_t_pad(const float* __restrict__ src,
                                                 bf16_t* __restrict__ dst,
                                                 int K, int N, int ldd,
                                                 size_t total) {
  for (size_t i = (size_t)blockIdx.x * 256 + threadIdx.x; i < total;
       i += (size_t)gridDim.x * 256) {
    int n = (int)(i / ldd), k = (int)(i % ldd);
    dst[i] = (k < K) ? (bf16_t)src[(size_t)k * N + n] : (bf16_t)0.0f;
  }
}

__global__ __launch_bounds__(256) void zero_bf16(bf16_t* __restrict__ dst,
                                                 size_t total) {
  for (size_t i = (size_t)blockIdx.x * 256 + threadIdx.x; i < total;
       i += (size_t)gridDim.x * 256)
    dst[i] = (bf16_t)0.0f;
}

// In-place softmax over one 4096-wide bf16 row per block (256 threads).
__global__ __launch_bounds__(256) void softmax_row(bf16_t* __restrict__ attn) {
  __shared__ float red[256];
  const int tid = threadIdx.x;
  bf16_t* p = attn + (size_t)blockIdx.x * MTOK;
  float v[16];
  float mx = -3.0e38f;
  #pragma unroll
  for (int i = 0; i < 16; ++i) {
    v[i] = (float)p[tid + (i << 8)];
    mx = fmaxf(mx, v[i]);
  }
  red[tid] = mx; __syncthreads();
  for (int s = 128; s > 0; s >>= 1) {
    if (tid < s) red[tid] = fmaxf(red[tid], red[tid + s]);
    __syncthreads();
  }
  mx = red[0]; __syncthreads();
  float sum = 0.0f;
  #pragma unroll
  for (int i = 0; i < 16; ++i) { v[i] = __expf(v[i] - mx); sum += v[i]; }
  red[tid] = sum; __syncthreads();
  for (int s = 128; s > 0; s >>= 1) {
    if (tid < s) red[tid] += red[tid + s];
    __syncthreads();
  }
  const float inv = 1.0f / red[0];
  #pragma unroll
  for (int i = 0; i < 16; ++i) p[tid + (i << 8)] = (bf16_t)(v[i] * inv);
}

// LayerNorm over one 3584-wide f32 row per block; writes bf16 y.
__global__ __launch_bounds__(256) void layernorm_row(
    const float* __restrict__ x, const float* __restrict__ g,
    const float* __restrict__ b, bf16_t* __restrict__ y) {
  __shared__ float red[256];
  const int tid = threadIdx.x;
  const float* px = x + (size_t)blockIdx.x * LDIM;
  float v[14];
  float s = 0.0f;
  #pragma unroll
  for (int i = 0; i < 14; ++i) { v[i] = px[tid + (i << 8)]; s += v[i]; }
  red[tid] = s; __syncthreads();
  for (int st = 128; st > 0; st >>= 1) {
    if (tid < st) red[tid] += red[tid + st];
    __syncthreads();
  }
  const float mu = red[0] * (1.0f / LDIM); __syncthreads();
  float s2 = 0.0f;
  #pragma unroll
  for (int i = 0; i < 14; ++i) { float d = v[i] - mu; s2 += d * d; }
  red[tid] = s2; __syncthreads();
  for (int st = 128; st > 0; st >>= 1) {
    if (tid < st) red[tid] += red[tid + st];
    __syncthreads();
  }
  const float rstd = rsqrtf(red[0] * (1.0f / LDIM) + 1e-5f);
  bf16_t* py = y + (size_t)blockIdx.x * LDIM;
  #pragma unroll
  for (int i = 0; i < 14; ++i) {
    int c = tid + (i << 8);
    py[c] = (bf16_t)((v[i] - mu) * rstd * g[c] + b[c]);
  }
}

inline unsigned gblocks(size_t total) {
  size_t b = (total + 255) / 256;
  return (unsigned)(b > 65535 ? 65535 : b);
}

}  // namespace

extern "C" void kernel_launch(void* const* d_in, const int* in_sizes, int n_in,
                              void* d_out, int out_size, void* d_ws,
                              size_t ws_size, hipStream_t stream) {
  const float* target   = (const float*)d_in[0];
  const float* source   = (const float*)d_in[1];
  const float* value    = (const float*)d_in[2];
  const float* wq_w     = (const float*)d_in[3];
  const float* wq_b     = (const float*)d_in[4];
  const float* wk_w     = (const float*)d_in[5];
  const float* wk_b     = (const float*)d_in[6];
  const float* wv_w     = (const float*)d_in[7];
  const float* wv_b     = (const float*)d_in[8];
  const float* resid_w  = (const float*)d_in[9];
  const float* resid_b  = (const float*)d_in[10];
  const float* out_w    = (const float*)d_in[11];
  const float* out_b    = (const float*)d_in[12];
  const float* ln_g     = (const float*)d_in[13];
  const float* ln_b     = (const float*)d_in[14];
  (void)in_sizes; (void)n_in; (void)out_size; (void)ws_size;

  char* ws = (char*)d_ws;
  bf16_t* tA    = (bf16_t*)(ws + OFF_TA);
  bf16_t* sA    = (bf16_t*)(ws + OFF_SA);
  bf16_t* vA    = (bf16_t*)(ws + OFF_VA);
  bf16_t* wqT   = (bf16_t*)(ws + OFF_WQT);
  bf16_t* wkT   = (bf16_t*)(ws + OFF_WKT);
  bf16_t* wvT   = (bf16_t*)(ws + OFF_WVT);
  bf16_t* rsT   = (bf16_t*)(ws + OFF_RST);
  bf16_t* outT  = (bf16_t*)(ws + OFF_OUTT);
  bf16_t* Qp    = (bf16_t*)(ws + OFF_QP);
  bf16_t* Kp    = (bf16_t*)(ws + OFF_KP);
  bf16_t* attnb = (bf16_t*)(ws + OFF_ATTN);
  bf16_t* Vt    = (bf16_t*)(ws + OFF_VT);
  float*  xbuf  = (float*) (ws + OFF_X);
  bf16_t* ybuf  = (bf16_t*)(ws + OFF_Y);

  // ---- convert/transpose/pad operands to bf16 ----
  {
    size_t t;
    t = (size_t)NTOK * RPAD;
    cvt_pad<<<gblocks(t), 256, 0, stream>>>(target, tA, RDIM, RPAD, t);
    t = (size_t)MTOK * LDIM;
    cvt_pad<<<gblocks(t), 256, 0, stream>>>(source, sA, LDIM, LDIM, t);
    cvt_pad<<<gblocks(t), 256, 0, stream>>>(value, vA, LDIM, LDIM, t);
    t = (size_t)RDIM * RPAD;
    cvt_t_pad<<<gblocks(t), 256, 0, stream>>>(wq_w, wqT, RDIM, RDIM, RPAD, t);
    t = (size_t)RDIM * LDIM;
    cvt_t_pad<<<gblocks(t), 256, 0, stream>>>(wk_w, wkT, LDIM, RDIM, LDIM, t);
    t = (size_t)VHDIM * LDIM;
    for (int h = 0; h < NHEAD; ++h)
      cvt_t_pad<<<gblocks(t), 256, 0, stream>>>(
          wv_w + (size_t)h * LDIM * VHDIM, wvT + (size_t)h * VHDIM * LDIM,
          LDIM, VHDIM, LDIM, t);
    t = (size_t)LDIM * RPAD;
    cvt_t_pad<<<gblocks(t), 256, 0, stream>>>(resid_w, rsT, RDIM, LDIM, RPAD, t);
    t = (size_t)LDIM * LDIM;
    cvt_t_pad<<<gblocks(t), 256, 0, stream>>>(out_w, outT, LDIM, LDIM, LDIM, t);
    // zero padded Q/K (Qp and Kp are contiguous)
    t = (size_t)(NTOK + MTOK) * QKLD;
    zero_bf16<<<gblocks(t), 256, 0, stream>>>(Qp, t);
  }

  const float inv_sqrt_hd = 1.0f / 6.0f;  // 1/sqrt(36)

  // Block covers 8 waves * MT*16 rows = 256 rows (MT=2). All row counts = 4096.
  constexpr int GR = NTOK / 256;  // 16 row-blocks

  // ---- Q projection (scaled by 1/sqrt(hd)) and K projection, padded stores --
  gemm_wmma<2, 1, 3><<<dim3(GR, RDIM / 16), 256, 0, stream>>>(
      tA, wqT, wq_b, Qp, RPAD, RPAD, RPAD, QKLD, inv_sqrt_hd, 0);
  gemm_wmma<2, 1, 3><<<dim3(GR, RDIM / 16), 256, 0, stream>>>(
      sA, wkT, wk_b, Kp, LDIM, LDIM, LDIM, QKLD, 1.0f, 0);

  // ---- V projections, stored transposed: Vt[h][d][m] ----
  for (int h = 0; h < NHEAD; ++h)
    gemm_wmma<2, 4, 2><<<dim3(GR, VHDIM / 64), 256, 0, stream>>>(
        vA, wvT + (size_t)h * VHDIM * LDIM, wv_b + (size_t)h * VHDIM,
        Vt + (size_t)h * VHDIM * MTOK, LDIM, LDIM, LDIM, MTOK, 1.0f, 0);

  // ---- residual projection writes x ----
  gemm_wmma<2, 4, 0><<<dim3(GR, LDIM / 64), 256, 0, stream>>>(
      tA, rsT, resid_b, xbuf, RPAD, RPAD, RPAD, LDIM, 1.0f, 0);

  // ---- per-head: scores -> softmax -> ctx accumulated into x ----
  for (int h = 0; h < NHEAD; ++h) {
    gemm_wmma<2, 4, 1><<<dim3(GR, MTOK / 64), 256, 0, stream>>>(
        Qp + h * HPAD, Kp + h * HPAD, nullptr, attnb,
        HPAD, QKLD, QKLD, MTOK, 1.0f, 0);
    softmax_row<<<NTOK, 256, 0, stream>>>(attnb);
    gemm_wmma<2, 4, 0><<<dim3(GR, VHDIM / 64), 256, 0, stream>>>(
        attnb, Vt + (size_t)h * VHDIM * MTOK, nullptr,
        xbuf + (size_t)h * VHDIM, MTOK, MTOK, MTOK, LDIM, 1.0f, 1);
  }

  // ---- LayerNorm -> bf16 y ----
  layernorm_row<<<NTOK, 256, 0, stream>>>(xbuf, ln_g, ln_b, ybuf);

  // ---- output projection -> d_out (f32) ----
  gemm_wmma<2, 4, 0><<<dim3(GR, LDIM / 64), 256, 0, stream>>>(
      ybuf, outT, out_b, (float*)d_out, LDIM, LDIM, LDIM, LDIM, 1.0f, 0);
}
